// WESS_Encoder_57123065037160
// MI455X (gfx1250) — compile-verified, matrix operand order checked by reading, not run on hardware
//
#include <hip/hip_runtime.h>
#include <hip/hip_bf16.h>

typedef __bf16 bf16_t;
typedef __attribute__((ext_vector_type(16))) __bf16 bf16x16;
typedef __attribute__((ext_vector_type(8)))  float   floatx8;

#define B_    16
#define T_    2048
#define NW_   256
#define L_    8
#define E_    256
#define HG_   768
#define HB_   256
#define NH_   4
#define DK_   64
#define NL_   2

// ---------------------------------------------------------------- helpers

__device__ __forceinline__ floatx8 zero8() {
  floatx8 z = {0.f, 0.f, 0.f, 0.f, 0.f, 0.f, 0.f, 0.f};
  return z;
}

__device__ __forceinline__ floatx8 wmma_bb(bf16x16 a, bf16x16 b, floatx8 c) {
  // D = A(16x32 bf16) x B(32x16 bf16) + C(16x16 f32)
  return __builtin_amdgcn_wmma_f32_16x16x32_bf16(false, a, false, b, (short)0, c,
                                                 false, false);
}

// A/B fragment loader for 16-bit 16x32 operands (wave32 layout):
//   lane l: row = l%16 ; hi = l/16
//   VGPR0-3 : K = kbase + hi*8 + 0..7   (16 contiguous bytes)
//   VGPR4-7 : K = kbase + 16 + hi*8 + 0..7
__device__ __forceinline__ bf16x16 load_frag(const bf16_t* base, int stride, int kbase) {
  int lane = threadIdx.x & 31;
  const bf16_t* p = base + (size_t)(lane & 15) * stride + kbase + ((lane >> 4) << 3);
  union { bf16x16 v; uint4 q[2]; } f;
  f.q[0] = *(const uint4*)(p);
  f.q[1] = *(const uint4*)(p + 16);
  return f.v;
}

// CDNA5 async global->LDS copy (16B per lane), ASYNCcnt-tracked.
// Generic shared pointers carry the LDS byte address in addr[31:0] (aperture rule),
// which is exactly what the VDST operand of the async op wants.
__device__ __forceinline__ void async_b128(const bf16_t* g, bf16_t* l) {
  unsigned int lds = (unsigned int)(unsigned long long)l;
  asm volatile("global_load_async_to_lds_b128 %0, %1, off"
               :: "v"(lds), "v"(g)
               : "memory");
}
__device__ __forceinline__ void async_wait0() {
  asm volatile("s_wait_asynccnt 0x0" ::: "memory");
}

__device__ __forceinline__ float sigmoidf_(float x) { return 1.f / (1.f + __expf(-x)); }

// ---------------------------------------------------------------- small kernels

__global__ void k_cast(const float* __restrict__ s, bf16_t* __restrict__ d, int n) {
  int i = blockIdx.x * blockDim.x + threadIdx.x;
  if (i < n) d[i] = (bf16_t)s[i];
}

// x_emb output: d_out tail region, (B,T,E) f32 gather
__global__ void k_emb_out(const int* __restrict__ x, const float* __restrict__ emb,
                          float* __restrict__ o) {
  int i = blockIdx.x * blockDim.x + threadIdx.x;
  if (i < B_ * T_ * E_) {
    int bt = i >> 8;
    o[i] = emb[x[bt] * E_ + (i & 255)];
  }
}

// h0 = bert + gru_fwd + gru_bwd  -> bf16 (GEMM A input)
__global__ void k_h0(const float* __restrict__ bert, const float* __restrict__ gp,
                     bf16_t* __restrict__ h0, int n) {
  int i = blockIdx.x * blockDim.x + threadIdx.x;
  if (i < n) h0[i] = (bf16_t)(bert[i] + gp[i] + gp[n + i]);
}

// h += sinusoid_table[w][e]  over (B,NW,E), f32 residual stream
__global__ void k_addpos(float* __restrict__ h) {
  int i = blockIdx.x * blockDim.x + threadIdx.x;
  if (i < B_ * NW_ * E_) {
    int w = (i >> 8) & (NW_ - 1);
    int e = i & 255;
    float expo = (float)(2 * (e >> 1)) * (1.0f / (float)E_);
    float ang = (float)w / __powf(10000.f, expo);
    h[i] += (e & 1) ? __cosf(ang) : __sinf(ang);
  }
}

// LayerNorm over last dim (=256), f32 in -> bf16 out (GEMM A input)
__global__ void __launch_bounds__(256) k_ln(const float* __restrict__ X,
                                            const float* __restrict__ g,
                                            const float* __restrict__ be,
                                            bf16_t* __restrict__ Y) {
  __shared__ float red[256];
  int row = blockIdx.x, c = threadIdx.x;
  float v = X[(size_t)row * HB_ + c];
  red[c] = v;
  __syncthreads();
  for (int s = 128; s > 0; s >>= 1) { if (c < s) red[c] += red[c + s]; __syncthreads(); }
  float mean = red[0] * (1.0f / HB_);
  __syncthreads();
  float d = v - mean;
  red[c] = d * d;
  __syncthreads();
  for (int s = 128; s > 0; s >>= 1) { if (c < s) red[c] += red[c + s]; __syncthreads(); }
  float var = red[0] * (1.0f / HB_);
  Y[(size_t)row * HB_ + c] = (bf16_t)(g[c] * d * rsqrtf(var + 1e-6f) + be[c]);
}

// ---------------------------------------------------------------- generic WMMA GEMM
// epilogue( sum_k A[m][k]*W[n][k] + bias[n] ), A bf16 row-major (async-staged to
// LDS), W bf16 row-major (NxK) fragments direct from global (L2 resident).
// 32-row M tile per WG; 128 threads = 4 waves, one 16-col tile per wave, B-frag
// reused across the two m-halves. k-loop software-pipelined: next iteration's
// fragments are issued before current WMMAs so waits stay partial.
__global__ void __launch_bounds__(128)
k_gemm(const bf16_t* __restrict__ A, int lda, const bf16_t* __restrict__ W, int ldw,
       const float* __restrict__ bias, const float* __restrict__ resid,
       float* __restrict__ C, bf16_t* __restrict__ Cb, int ldc, int K, int act,
       bf16_t* __restrict__ vt) {
  extern __shared__ char smem[];
  bf16_t* As = (bf16_t*)smem;  // 32 x K bf16
  int tid = threadIdx.x, wave = tid >> 5, lane = tid & 31;
  int mt = blockIdx.x;
  const bf16_t* Arow = A + (size_t)mt * 32 * lda;
  int chunks = (32 * K) >> 3;  // 16B chunks
  for (int i = tid; i < chunks; i += 128) {
    int e = i << 3;
    int r = e / K, c2 = e - r * K;
    async_b128(Arow + (size_t)r * lda + c2, As + r * K + c2);
  }
  async_wait0();
  __syncthreads();

  int n0 = (blockIdx.y * 4 + wave) * 16;
  const bf16_t* Wb = W + (size_t)n0 * ldw;
  floatx8 acc0 = zero8(), acc1 = zero8();
  bf16x16 a0 = load_frag(As, K, 0);
  bf16x16 a1 = load_frag(As + 16 * K, K, 0);
  bf16x16 b0 = load_frag(Wb, ldw, 0);
  for (int k = 32; k < K; k += 32) {
    bf16x16 a0n = load_frag(As, K, k);
    bf16x16 a1n = load_frag(As + 16 * K, K, k);
    bf16x16 b0n = load_frag(Wb, ldw, k);
    acc0 = wmma_bb(a0, b0, acc0);
    acc1 = wmma_bb(a1, b0, acc1);
    a0 = a0n; a1 = a1n; b0 = b0n;
  }
  acc0 = wmma_bb(a0, b0, acc0);
  acc1 = wmma_bb(a1, b0, acc1);

  int col = n0 + (lane & 15);
  int hi = lane >> 4;
  float bv = bias ? bias[col] : 0.f;
#pragma unroll
  for (int half = 0; half < 2; ++half) {
    floatx8 acc = half ? acc1 : acc0;
#pragma unroll
    for (int i = 0; i < 8; ++i) {
      int row = mt * 32 + half * 16 + i + 8 * hi;
      float v = acc[i] + bv;
      if (act == 1) {
        v = v > 0.f ? v : 0.f;
      } else if (act == 2) {
        float x3 = v * v * v;
        v = 0.5f * v * (1.0f + tanhf(0.7978845608f * (v + 0.044715f * x3)));
      }
      if (resid) v += resid[(size_t)row * ldc + col];
      if (C) C[(size_t)row * ldc + col] = v;
      if (Cb) Cb[(size_t)row * ldc + col] = (bf16_t)v;
      if (vt) {
        int b = row >> 8, q = row & 255;
        int h = col >> 6, d = col & 63;
        vt[((((size_t)b * NH_ + h) * DK_ + d) << 8) + q] = (bf16_t)v;
      }
    }
  }
}

// ---------------------------------------------------------------- BiGRU (recurrent, per-word-tile)
// grid=(B*NW/16, 2 dirs), 256 threads = 8 waves. Each WG owns 16 words for the full
// 2304-wide gate output; h lives in LDS across the 8 steps (bf16 A-copy + f32 copy).
// Each wave owns 6 hidden blocks of 16. Emits h(step0)+h(step7) to gp[dir].
__global__ void __launch_bounds__(256)
k_gru(const int* __restrict__ x, const int* __restrict__ gate,
      const float* __restrict__ emb,
      const bf16_t* __restrict__ WihF, const bf16_t* __restrict__ WhhF,
      const bf16_t* __restrict__ WihB, const bf16_t* __restrict__ WhhB,
      const float* __restrict__ bihF, const float* __restrict__ bhhF,
      const float* __restrict__ bihB, const float* __restrict__ bhhB,
      float* __restrict__ gp) {
  extern __shared__ char smem[];
  bf16_t* hbf = (bf16_t*)smem;                                 // 16 x 768 bf16
  float*  hf  = (float*)(smem + 16 * HG_ * 2);                 // 16 x 768 f32
  bf16_t* xbf = (bf16_t*)(smem + 16 * HG_ * 2 + 16 * HG_ * 4); // 16 x 256 bf16
  int* eidx   = (int*)(smem + 16 * HG_ * 2 + 16 * HG_ * 4 + 16 * E_ * 2);

  int dir = blockIdx.y;
  const bf16_t* Wih = dir ? WihB : WihF;
  const bf16_t* Whh = dir ? WhhB : WhhF;
  const float*  bih = dir ? bihB : bihF;
  const float*  bhh = dir ? bhhB : bhhF;

  int tid = threadIdx.x, lane = tid & 31, wave = tid >> 5;
  int hi = lane >> 4;
  int wt = blockIdx.x;

  for (int i = tid; i < 16 * HG_; i += 256) { hf[i] = 0.f; hbf[i] = (bf16_t)0.f; }

  float femb[6][8];
  float hnv[6][8];

  for (int step = 0; step < L_; ++step) {
    int t = dir ? (L_ - 1 - step) : step;
    if (tid < 16) {
      int wg = wt * 16 + tid;
      int b = wg >> 8, w = wg & 255;
      int start = gate[b * (NW_ + 1) + w];
      eidx[tid] = x[b * T_ + start + t];
    }
    __syncthreads();
    for (int i = tid; i < 16 * E_; i += 256) {
      int r = i >> 8;
      xbf[i] = (bf16_t)emb[(size_t)eidx[r] * E_ + (i & 255)];
    }
    __syncthreads();

#pragma unroll
    for (int blk = 0; blk < 6; ++blk) {
      int n0 = (wave * 6 + blk) * 16;
      floatx8 ar = zero8(), az = zero8(), angi = zero8(), angh = zero8();
      const bf16_t* Wr = Wih + (size_t)n0 * E_;
      const bf16_t* Wz = Wih + (size_t)(HG_ + n0) * E_;
      const bf16_t* Wn = Wih + (size_t)(2 * HG_ + n0) * E_;
      for (int k = 0; k < E_; k += 32) {
        bf16x16 a = load_frag(xbf, E_, k);
        ar   = wmma_bb(a, load_frag(Wr, E_, k), ar);
        az   = wmma_bb(a, load_frag(Wz, E_, k), az);
        angi = wmma_bb(a, load_frag(Wn, E_, k), angi);
      }
      const bf16_t* Ur = Whh + (size_t)n0 * HG_;
      const bf16_t* Uz = Whh + (size_t)(HG_ + n0) * HG_;
      const bf16_t* Un = Whh + (size_t)(2 * HG_ + n0) * HG_;
      for (int k = 0; k < HG_; k += 32) {
        bf16x16 a = load_frag(hbf, HG_, k);
        ar   = wmma_bb(a, load_frag(Ur, HG_, k), ar);
        az   = wmma_bb(a, load_frag(Uz, HG_, k), az);
        angh = wmma_bb(a, load_frag(Un, HG_, k), angh);
      }
      int col = n0 + (lane & 15);
      float br = bih[col] + bhh[col];
      float bz = bih[HG_ + col] + bhh[HG_ + col];
      float bin = bih[2 * HG_ + col];
      float bhn = bhh[2 * HG_ + col];
#pragma unroll
      for (int i = 0; i < 8; ++i) {
        int row = i + 8 * hi;
        float r_ = sigmoidf_(ar[i] + br);
        float z_ = sigmoidf_(az[i] + bz);
        float n_ = tanhf(angi[i] + bin + r_ * (angh[i] + bhn));
        float ho = hf[row * HG_ + col];
        hnv[blk][i] = (1.f - z_) * n_ + z_ * ho;
      }
    }
    __syncthreads();  // all waves done reading h
#pragma unroll
    for (int blk = 0; blk < 6; ++blk) {
      int col = (wave * 6 + blk) * 16 + (lane & 15);
#pragma unroll
      for (int i = 0; i < 8; ++i) {
        int row = i + 8 * hi;
        float v = hnv[blk][i];
        hf[row * HG_ + col] = v;
        hbf[row * HG_ + col] = (bf16_t)v;
        if (step == 0) femb[blk][i] = v;
        else if (step == L_ - 1) femb[blk][i] += v;
      }
    }
    __syncthreads();
  }

  float* out = gp + (size_t)dir * (B_ * NW_ * HG_);
#pragma unroll
  for (int blk = 0; blk < 6; ++blk) {
    int col = (wave * 6 + blk) * 16 + (lane & 15);
#pragma unroll
    for (int i = 0; i < 8; ++i) {
      int row = wt * 16 + i + 8 * hi;
      out[(size_t)row * HG_ + col] = femb[blk][i];
    }
  }
}

// ---------------------------------------------------------------- attention
// grid = B*NH*(NW/16), 128 threads = 4 waves. Per WG: one 16-query tile of one head.
// Q/K tiles async-staged (bf16). scores via WMMA -> parallel softmax (8 lanes/row)
// -> P x V^T via WMMA -> O bf16 (A input of the Wo GEMM).
__global__ void __launch_bounds__(128)
k_attn(const bf16_t* __restrict__ Q, const bf16_t* __restrict__ Kx,
       const bf16_t* __restrict__ Vt, bf16_t* __restrict__ O) {
  __shared__ __align__(16) bf16_t qs[16 * DK_];
  __shared__ __align__(16) bf16_t ks[4][16 * DK_];
  __shared__ __align__(16) float  sc[16 * NW_];
  __shared__ __align__(16) bf16_t ps[16 * NW_];
  __shared__ float red2[16][8];
  int tid = threadIdx.x, lane = tid & 31, wave = tid >> 5;
  int qt = blockIdx.x & 15;
  int bh = blockIdx.x >> 4;
  int b = bh >> 2, h = bh & 3;

  {  // stage Q tile: 128 chunks of 16B, one per thread
    int e = tid << 3;
    int r = e >> 6, c = e & 63;
    async_b128(Q + (size_t)(b * NW_ + qt * 16 + r) * HB_ + h * DK_ + c, qs + e);
    async_wait0();
  }
  __syncthreads();

  for (int kt = wave * 4; kt < wave * 4 + 4; ++kt) {
    for (int i = lane; i < 128; i += 32) {  // per-wave K tile staging
      int e = i << 3;
      int r = e >> 6, c = e & 63;
      async_b128(Kx + (size_t)(b * NW_ + kt * 16 + r) * HB_ + h * DK_ + c,
                 ks[wave] + e);
    }
    async_wait0();
    floatx8 acc = zero8();
    for (int k = 0; k < DK_; k += 32)
      acc = wmma_bb(load_frag(qs, DK_, k), load_frag(ks[wave], DK_, k), acc);
    int colk = kt * 16 + (lane & 15);
    int hi = lane >> 4;
#pragma unroll
    for (int i = 0; i < 8; ++i) sc[(i + 8 * hi) * NW_ + colk] = acc[i] * 0.125f;
  }
  __syncthreads();

  // softmax: 8 lanes per row, LDS tree reduction
  {
    int row = tid >> 3, sub = tid & 7;
    float mx = -1e30f;
    for (int j = sub; j < NW_; j += 8) mx = fmaxf(mx, sc[row * NW_ + j]);
    red2[row][sub] = mx;
    __syncthreads();
    if (sub < 4) red2[row][sub] = fmaxf(red2[row][sub], red2[row][sub + 4]);
    __syncthreads();
    if (sub < 2) red2[row][sub] = fmaxf(red2[row][sub], red2[row][sub + 2]);
    __syncthreads();
    if (sub < 1) red2[row][sub] = fmaxf(red2[row][sub], red2[row][sub + 1]);
    __syncthreads();
    mx = red2[row][0];
    __syncthreads();
    float s = 0.f;
    for (int j = sub; j < NW_; j += 8) {
      float e = __expf(sc[row * NW_ + j] - mx);
      sc[row * NW_ + j] = e;
      s += e;
    }
    red2[row][sub] = s;
    __syncthreads();
    if (sub < 4) red2[row][sub] += red2[row][sub + 4];
    __syncthreads();
    if (sub < 2) red2[row][sub] += red2[row][sub + 2];
    __syncthreads();
    if (sub < 1) red2[row][sub] += red2[row][sub + 1];
    __syncthreads();
    float inv = 1.0f / red2[row][0];
    for (int j = sub; j < NW_; j += 8)
      ps[row * NW_ + j] = (bf16_t)(sc[row * NW_ + j] * inv);
  }
  __syncthreads();

  {
    int n0 = wave * 16;
    const bf16_t* Wb = Vt + ((size_t)bh * DK_ + n0) * NW_;
    floatx8 acc = zero8();
    for (int k = 0; k < NW_; k += 32)
      acc = wmma_bb(load_frag(ps, NW_, k), load_frag(Wb, NW_, k), acc);
    int hi = lane >> 4, d = n0 + (lane & 15);
#pragma unroll
    for (int i = 0; i < 8; ++i) {
      int q = qt * 16 + i + 8 * hi;
      O[(size_t)(b * NW_ + q) * HB_ + h * DK_ + d] = (bf16_t)acc[i];
    }
  }
}

// ---------------------------------------------------------------- launch

extern "C" void kernel_launch(void* const* d_in, const int* in_sizes, int n_in,
                              void* d_out, int out_size, void* d_ws, size_t ws_size,
                              hipStream_t stream) {
  (void)in_sizes; (void)n_in; (void)out_size; (void)ws_size;

  const int*   xi     = (const int*)d_in[0];
  const float* bert   = (const float*)d_in[1];
  const int*   gate   = (const int*)d_in[2];
  const float* emb    = (const float*)d_in[4];
  const float* whh_b  = (const float*)d_in[5];
  const float* whh_f  = (const float*)d_in[6];
  const float* wih_b  = (const float*)d_in[7];
  const float* wih_f  = (const float*)d_in[8];
  const float* bhh_b  = (const float*)d_in[9];
  const float* bhh_f  = (const float*)d_in[10];
  const float* bih_b  = (const float*)d_in[11];
  const float* bih_f  = (const float*)d_in[12];
  const float* pre_W1 = (const float*)d_in[45];
  const float* pre_W2 = (const float*)d_in[46];
  const float* pre_b1 = (const float*)d_in[47];
  const float* pre_b2 = (const float*)d_in[48];

  // ---- workspace layout: bf16 region first, f32 region after
  bf16_t* wb = (bf16_t*)d_ws;
  size_t wo = 0;
  auto walloc = [&](size_t n) { size_t r = wo; wo += n; return r; };
  size_t oWHHB = walloc(2304 * 768), oWHHF = walloc(2304 * 768);
  size_t oWIHB = walloc(2304 * 256), oWIHF = walloc(2304 * 256);
  size_t oPW1 = walloc(1024 * 768), oPW2 = walloc(256 * 1024);
  size_t oLW1[NL_], oLW2[NL_], oLWk[NL_], oLWo[NL_], oLWq[NL_], oLWv[NL_];
  for (int l = 0; l < NL_; ++l) {
    oLW1[l] = walloc(1024 * 256);
    oLW2[l] = walloc(256 * 1024);
    oLWk[l] = walloc(256 * 256);
    oLWo[l] = walloc(256 * 256);
    oLWq[l] = walloc(256 * 256);
    oLWv[l] = walloc(256 * 256);
  }
  size_t n768 = (size_t)B_ * NW_ * HG_;  // 3,145,728
  size_t n256 = (size_t)B_ * NW_ * HB_;  // 1,048,576
  size_t oVT  = walloc((size_t)B_ * NH_ * DK_ * NW_);  // transposed V
  size_t oH0B = walloc(n768);                          // bf16 pre-net input
  size_t oF1B = walloc((size_t)4096 * 1024);           // bf16 pre/FFN intermediate
  size_t oABB = walloc(n256);                          // bf16 LN output
  size_t oQBB = walloc(n256);
  size_t oKBB = walloc(n256);
  size_t oOBB = walloc(n256);

  size_t fbase = (wo * 2 + 255) & ~(size_t)255;
  float* fw = (float*)((char*)d_ws + fbase);
  float* gp = fw;              // [2][B*NW][HG]
  float* hb = gp + 2 * n768;   // f32 residual stream (4096 x 256)

  auto cast = [&](const float* s, size_t off, int n) {
    k_cast<<<dim3((n + 255) / 256), 256, 0, stream>>>(s, wb + off, n);
  };
  cast(whh_b, oWHHB, 2304 * 768);
  cast(whh_f, oWHHF, 2304 * 768);
  cast(wih_b, oWIHB, 2304 * 256);
  cast(wih_f, oWIHF, 2304 * 256);
  cast(pre_W1, oPW1, 1024 * 768);
  cast(pre_W2, oPW2, 256 * 1024);
  for (int l = 0; l < NL_; ++l) {
    int bs = 13 + 16 * l;  // layer leaves: W1 W2 Wk Wo Wq Wv b1 b2 be1 be2 bk bo bq bv g1 g2
    cast((const float*)d_in[bs + 0], oLW1[l], 1024 * 256);
    cast((const float*)d_in[bs + 1], oLW2[l], 256 * 1024);
    cast((const float*)d_in[bs + 2], oLWk[l], 256 * 256);
    cast((const float*)d_in[bs + 3], oLWo[l], 256 * 256);
    cast((const float*)d_in[bs + 4], oLWq[l], 256 * 256);
    cast((const float*)d_in[bs + 5], oLWv[l], 256 * 256);
  }

  // x_emb output (second tuple element)
  k_emb_out<<<dim3((B_ * T_ * E_ + 255) / 256), 256, 0, stream>>>(
      xi, emb, (float*)d_out + n256);

  // BiGRU: 256 word tiles x 2 directions
  size_t gru_smem = 16 * HG_ * 2 + 16 * HG_ * 4 + 16 * E_ * 2 + 64;
  k_gru<<<dim3(B_ * NW_ / 16, 2), 256, gru_smem, stream>>>(
      xi, gate, emb, wb + oWIHF, wb + oWHHF, wb + oWIHB, wb + oWHHB,
      bih_f, bhh_f, bih_b, bhh_b, gp);

  k_h0<<<dim3((int)((n768 + 255) / 256)), 256, 0, stream>>>(bert, gp, wb + oH0B,
                                                            (int)n768);

  auto gemm = [&](const bf16_t* Ap, int K, size_t oW, const float* bias,
                  const float* resid, float* Cp, bf16_t* Cbp, int N, int act,
                  bf16_t* vt) {
    k_gemm<<<dim3(B_ * NW_ / 32, N / 64), 128, (size_t)32 * K * 2, stream>>>(
        Ap, K, wb + oW, K, bias, resid, Cp, Cbp, N, K, act, vt);
  };

  // pre-net
  gemm(wb + oH0B, 768, oPW1, pre_b1, nullptr, nullptr, wb + oF1B, 1024, 1, nullptr);
  gemm(wb + oF1B, 1024, oPW2, pre_b2, nullptr, hb, nullptr, 256, 0, nullptr);
  k_addpos<<<dim3((int)(n256 / 256)), 256, 0, stream>>>(hb);

  // transformer layers
  for (int l = 0; l < NL_; ++l) {
    int bs = 13 + 16 * l;
    const float* b1  = (const float*)d_in[bs + 6];
    const float* b2  = (const float*)d_in[bs + 7];
    const float* be1 = (const float*)d_in[bs + 8];
    const float* be2 = (const float*)d_in[bs + 9];
    const float* bk  = (const float*)d_in[bs + 10];
    const float* bo  = (const float*)d_in[bs + 11];
    const float* bq  = (const float*)d_in[bs + 12];
    const float* bv  = (const float*)d_in[bs + 13];
    const float* g1  = (const float*)d_in[bs + 14];
    const float* g2  = (const float*)d_in[bs + 15];

    k_ln<<<dim3(B_ * NW_), 256, 0, stream>>>(hb, g1, be1, wb + oABB);
    gemm(wb + oABB, 256, oLWq[l], bq, nullptr, nullptr, wb + oQBB, 256, 0, nullptr);
    gemm(wb + oABB, 256, oLWk[l], bk, nullptr, nullptr, wb + oKBB, 256, 0, nullptr);
    gemm(wb + oABB, 256, oLWv[l], bv, nullptr, nullptr, nullptr, 256, 0, wb + oVT);
    k_attn<<<dim3(B_ * NH_ * (NW_ / 16)), 128, 0, stream>>>(
        wb + oQBB, wb + oKBB, wb + oVT, wb + oOBB);
    gemm(wb + oOBB, 256, oLWo[l], bo, hb, hb, nullptr, 256, 0, nullptr);
    k_ln<<<dim3(B_ * NW_), 256, 0, stream>>>(hb, g2, be2, wb + oABB);
    gemm(wb + oABB, 256, oLW1[l], b1, nullptr, nullptr, wb + oF1B, 1024, 2, nullptr);
    float* outp = (l == NL_ - 1) ? (float*)d_out : hb;
    gemm(wb + oF1B, 1024, oLW2[l], b2, hb, outp, nullptr, 256, 0, nullptr);
  }
}
